// FastSpeech2MIDILess_32598801777236
// MI455X (gfx1250) — matrix operand order repaired and checked by reading, not compiled, not run on hardware
//
#include <hip/hip_runtime.h>
#include <math.h>

typedef __attribute__((ext_vector_type(16))) __bf16 v16bf;
typedef __attribute__((ext_vector_type(8)))  __bf16 bf16x8;
typedef __attribute__((ext_vector_type(8)))  float   v8f;
typedef __attribute__((ext_vector_type(8)))  unsigned short us8;

union Frag16 { v16bf v; bf16x8 h[2]; };

__device__ __forceinline__ unsigned short f2bf(float f) {
  unsigned u = __float_as_uint(f);
  unsigned r = (u + 0x7FFFu + ((u >> 16) & 1u)) >> 16;
  return (unsigned short)r;
}

__device__ __forceinline__ v8f wmma_bf16(v16bf a, v16bf b, v8f c) {
  // D = A(16x32 bf16) * B(32x16 bf16) + C(16x16 f32)
  return __builtin_amdgcn_wmma_f32_16x16x32_bf16(false, a, false, b, (short)0, c, false, false);
}

__device__ __forceinline__ v8f vzero8() {
  v8f z;
#pragma unroll
  for (int i = 0; i < 8; i++) z[i] = 0.f;
  return z;
}

// Async global->LDS copy of 16 bytes per lane (CDNA5 ASYNCcnt path).
// ldsoff: 32-bit LDS byte address (low 32 bits of the flat shared-pointer).
__device__ __forceinline__ void async_copy_b128(unsigned ldsoff, const void* gaddr) {
  asm volatile("global_load_async_to_lds_b128 %0, %1, off"
               :: "v"(ldsoff), "v"(gaddr) : "memory");
}
__device__ __forceinline__ void async_wait0() {
  asm volatile("s_wait_asynccnt 0x0" ::: "memory");
}
__device__ __forceinline__ unsigned lds_addr32(const void* p) {
  return (unsigned)(unsigned long long)p;  // low 32 bits = LDS offset in aperture
}

// ---------------------------------------------------------------- f32 -> bf16
__global__ void k_f32_to_bf16(const float* __restrict__ in, unsigned short* __restrict__ out, int n) {
  int i = blockIdx.x * blockDim.x + threadIdx.x;
  if (i < n) out[i] = f2bf(in[i]);
}

// ---------------------------------------------------------------- length regulator
__global__ __launch_bounds__(512) void k_mel2ph(const int* __restrict__ tokens,
                                                const int* __restrict__ dur,
                                                int* __restrict__ mel2ph) {
  __shared__ int cs[512];
  int b = blockIdx.x, tid = threadIdx.x;
  int d = dur[b * 512 + tid];
  if (tokens[b * 512 + tid] == 0) d = 0;
  cs[tid] = d;
  for (int off = 1; off < 512; off <<= 1) {
    __syncthreads();
    int v = (tid >= off) ? cs[tid - off] : 0;
    __syncthreads();
    cs[tid] += v;
  }
  __syncthreads();
  int total = cs[511];
  for (int fr = tid; fr < 4096; fr += 512) {
    int res = 0;
    if (fr < total) {
      int lo = 0, hi = 511;
      while (lo < hi) { int mid = (lo + hi) >> 1; if (cs[mid] > fr) hi = mid; else lo = mid + 1; }
      res = lo + 1;
    }
    mel2ph[b * 4096 + fr] = res;
  }
}

// ---------------------------------------------------------------- embedding + sinusoid
__global__ __launch_bounds__(256) void k_embed(const int* __restrict__ tokens,
                                               const int* __restrict__ dur,
                                               const float* __restrict__ txt_w,
                                               const float* __restrict__ dur_w,
                                               const float* __restrict__ dur_b,
                                               float* __restrict__ x) {
  int row = blockIdx.x;       // b*512 + t
  int h = threadIdx.x;
  int t = row & 511;
  int tok = tokens[row];
  float nonpad = (tok != 0) ? 1.f : 0.f;
  float durf = (tok != 0) ? (float)dur[row] : 0.f;
  float j = (float)(h & ~1);
  float div = __expf(j * (-9.210340371976184f / 256.0f));
  float ang = (float)t * div;
  float pos = (h & 1) ? __cosf(ang) : __sinf(ang);
  float v = txt_w[tok * 256 + h] * 16.0f + durf * dur_w[h] + dur_b[h] + pos;
  x[(size_t)row * 256 + h] = v * nonpad;
}

// ---------------------------------------------------------------- LayerNorm (wave per row)
template <bool OUTBF16, bool MASK>
__global__ __launch_bounds__(256) void k_ln(const float* __restrict__ x,
                                            const float* __restrict__ sc,
                                            const float* __restrict__ bi,
                                            const int* __restrict__ tokens,
                                            unsigned short* __restrict__ outB,
                                            float* __restrict__ outF) {
  int wave = threadIdx.x >> 5, lane = threadIdx.x & 31;
  int row = blockIdx.x * 8 + wave;
  const float* xr = x + (size_t)row * 256;
  float v[8];
  float s = 0.f;
#pragma unroll
  for (int i = 0; i < 8; i++) { v[i] = xr[lane + 32 * i]; s += v[i]; }
#pragma unroll
  for (int m = 16; m >= 1; m >>= 1) s += __shfl_xor(s, m, 32);
  float mu = s * (1.f / 256.f);
  float var = 0.f;
#pragma unroll
  for (int i = 0; i < 8; i++) { float d = v[i] - mu; var += d * d; }
#pragma unroll
  for (int m = 16; m >= 1; m >>= 1) var += __shfl_xor(var, m, 32);
  float rs = rsqrtf(var * (1.f / 256.f) + 1e-5f);
  float msk = 1.f;
  if (MASK) msk = (tokens[row] != 0) ? 1.f : 0.f;
#pragma unroll
  for (int i = 0; i < 8; i++) {
    int e = lane + 32 * i;
    float o = (v[i] - mu) * rs * sc[e] + bi[e];
    if (MASK) o *= msk;
    if (OUTBF16) outB[(size_t)row * 256 + e] = f2bf(o);
    else         outF[(size_t)row * 256 + e] = o;
  }
}

// ---------------------------------------------------------------- tiled bf16 WMMA GEMM
// C[M,N] = A[M,K](bf16) * B[K,N](bf16) + bias, optional residual/relu/mask, M=8192.
// CONV: A is virtual im2col of h[b,t,c] with 9 taps (K = 9*256), SAME padding.
template <bool CONV, bool RELU, bool RESID, bool MASK, bool OUTBF16>
__global__ __launch_bounds__(256) void k_gemm(const unsigned short* __restrict__ A,
                                              const unsigned short* __restrict__ Bw,
                                              const float* __restrict__ bias,
                                              const float* __restrict__ resid,
                                              const int* __restrict__ tokens,
                                              unsigned short* __restrict__ outB,
                                              float* __restrict__ outF,
                                              int N, int K, int lda) {
  __shared__ __align__(16) unsigned short As[64 * 32];   // 64 rows x 32 k (bf16)
  __shared__ __align__(16) unsigned short Bst[64 * 32];  // transposed: 64 n x 32 k
  int tid = threadIdx.x;
  int wave = tid >> 5, lane = tid & 31;
  int mtile = wave & 3, npair = wave >> 2;
  int blockM = blockIdx.y * 64;
  int blockN = blockIdx.x * 64;

  int arow = tid >> 2;          // 0..63
  int acg = (tid & 3) * 8;      // 0,8,16,24
  int bk = tid >> 3;            // 0..31
  int bcg = (tid & 7) * 8;      // 0..56

  int grow = blockM + arow;
  int gb = grow >> 9;
  int gt = grow & 511;

  v8f acc0 = vzero8();
  v8f acc1 = vzero8();

  int m = lane & 15;
  int kbA = (lane >= 16) ? 16 : 0;
  int kbB = (lane >= 16) ? 16 : 0;
  int nb0 = (npair * 2 + 0) * 16 + m;
  int nb1 = (npair * 2 + 1) * 16 + m;

  unsigned ldsA = lds_addr32(&As[arow * 32 + acg]);

  for (int k0 = 0; k0 < K; k0 += 32) {
    __syncthreads();
    // ---- stage A tile
    if (CONV) {
      us8 aval;
#pragma unroll
      for (int i = 0; i < 8; i++) aval[i] = 0;
      int kk = k0 + acg;
      int tap = kk >> 8;
      int c = kk & 255;
      int st = gt + tap - 4;
      if (st >= 0 && st < 512)
        aval = *reinterpret_cast<const us8*>(A + ((size_t)(gb * 512 + st) * 256 + c));
      *reinterpret_cast<us8*>(&As[arow * 32 + acg]) = aval;
    } else {
      // CDNA5 async DMA: global -> LDS without a VGPR round-trip (ASYNCcnt).
      async_copy_b128(ldsA, A + ((size_t)grow * lda + k0 + acg));
    }
    // ---- stage B tile (transposed into LDS; needs per-element shuffle)
    {
      us8 bval = *reinterpret_cast<const us8*>(Bw + ((size_t)(k0 + bk) * N + blockN + bcg));
#pragma unroll
      for (int i = 0; i < 8; i++) Bst[(bcg + i) * 32 + bk] = bval[i];
      if (k0 + 32 < K)
        __builtin_prefetch(Bw + ((size_t)(k0 + 32 + bk) * N + blockN + bcg), 0, 0);
    }
    if (!CONV) async_wait0();  // our async writes landed; barrier covers other waves
    __syncthreads();
    // ---- fragments + WMMA
    const __bf16* asp = reinterpret_cast<const __bf16*>(As);
    const __bf16* bsp = reinterpret_cast<const __bf16*>(Bst);
    Frag16 fa;
    fa.h[0] = *reinterpret_cast<const bf16x8*>(asp + (mtile * 16 + m) * 32 + kbA);
    fa.h[1] = *reinterpret_cast<const bf16x8*>(asp + (mtile * 16 + m) * 32 + kbA + 8);
    Frag16 fb0, fb1;
    fb0.h[0] = *reinterpret_cast<const bf16x8*>(bsp + nb0 * 32 + kbB);
    fb0.h[1] = *reinterpret_cast<const bf16x8*>(bsp + nb0 * 32 + kbB + 8);
    fb1.h[0] = *reinterpret_cast<const bf16x8*>(bsp + nb1 * 32 + kbB);
    fb1.h[1] = *reinterpret_cast<const bf16x8*>(bsp + nb1 * 32 + kbB + 8);
    acc0 = wmma_bf16(fa.v, fb0.v, acc0);
    acc1 = wmma_bf16(fa.v, fb1.v, acc1);
  }

  // ---- epilogue
  int half = lane >> 4;
  int colid = lane & 15;
#pragma unroll
  for (int ti = 0; ti < 2; ti++) {
    v8f acc = ti ? acc1 : acc0;
    int col = blockN + (npair * 2 + ti) * 16 + colid;
    float bv = bias[col];
#pragma unroll
    for (int r = 0; r < 8; r++) {
      int row = blockM + mtile * 16 + r + 8 * half;
      float v = acc[r] + bv;
      if (RESID) v += resid[(size_t)row * N + col];
      if (RELU) v = fmaxf(v, 0.f);
      if (MASK) v *= (tokens[row] != 0) ? 1.f : 0.f;
      if (OUTBF16) outB[(size_t)row * N + col] = f2bf(v);
      else         outF[(size_t)row * N + col] = v;
    }
  }
}

// ---------------------------------------------------------------- attention
// grid = (T/32, NH, B), block = 64 (2 waves). Each wave owns 16 q-rows.
// Full 512-wide score row kept in registers (32 v8f accumulators / wave).
__global__ __launch_bounds__(64) void k_attn(const unsigned short* __restrict__ qkv,
                                             const int* __restrict__ tokens,
                                             unsigned short* __restrict__ obuf) {
  __shared__ __align__(16) unsigned short KV[64 * 128];  // K chunk (row-major) / V chunk (transposed)
  __shared__ __align__(16) unsigned short P[32 * 512];   // probs bf16
  __shared__ float padneg[512];
  int tid = threadIdx.x;
  int wave = tid >> 5, lane = tid & 31;
  int qtile = blockIdx.x;
  int head = blockIdx.y;
  int b = blockIdx.z;
  int m = lane & 15;
  int half = lane >> 4;
  int colid = lane & 15;
  int kbA = half ? 16 : 0;
  int kbB = half ? 16 : 0;

  // Q fragments (A-matrix 16x32 per k-step, 4 steps cover d=128)
  Frag16 qf[4];
  {
    const unsigned short* qp = qkv + ((size_t)(b * 512 + qtile * 32 + wave * 16 + m) * 768 + head * 128);
#pragma unroll
    for (int ks = 0; ks < 4; ks++) {
      qf[ks].h[0] = *reinterpret_cast<const bf16x8*>(qp + ks * 32 + kbA);
      qf[ks].h[1] = *reinterpret_cast<const bf16x8*>(qp + ks * 32 + kbA + 8);
    }
  }
  for (int i = tid; i < 512; i += 64)
    padneg[i] = (tokens[b * 512 + i] == 0) ? -1e9f : 0.f;

  unsigned ldsK = lds_addr32(&KV[tid * 128]);

  v8f accS[32];
#pragma unroll
  for (int t = 0; t < 32; t++) accS[t] = vzero8();

  // ---- phase 1: S = Q * K^T  (K chunk staged via async DMA, row-major copy)
#pragma unroll
  for (int ch = 0; ch < 8; ch++) {
    __syncthreads();
    {
      const unsigned short* kp = qkv + ((size_t)(b * 512 + ch * 64 + tid) * 768 + 256 + head * 128);
#pragma unroll
      for (int g = 0; g < 16; g++)
        async_copy_b128(ldsK + g * 16, kp + g * 8);
      async_wait0();
    }
    __syncthreads();
#pragma unroll
    for (int nt = 0; nt < 4; nt++) {
      int kcol = nt * 16 + colid;  // key row within chunk = this lane's B column
#pragma unroll
      for (int ks = 0; ks < 4; ks++) {
        Frag16 fb;
        const __bf16* kbp = reinterpret_cast<const __bf16*>(KV) + kcol * 128 + ks * 32 + kbB;
        fb.h[0] = *reinterpret_cast<const bf16x8*>(kbp);
        fb.h[1] = *reinterpret_cast<const bf16x8*>(kbp + 8);
        accS[ch * 4 + nt] = wmma_bf16(qf[ks].v, fb.v, accS[ch * 4 + nt]);
      }
    }
  }

  // ---- softmax (rows are spread over 16 lanes of each half-wave)
  const float SC = 0.08838834764831845f;  // 1/sqrt(128)
  float rm[8], rsum[8];
#pragma unroll
  for (int r = 0; r < 8; r++) rm[r] = -3.0e38f;
#pragma unroll
  for (int t = 0; t < 32; t++) {
    float pn = padneg[t * 16 + colid];
#pragma unroll
    for (int r = 0; r < 8; r++) {
      float v = accS[t][r] * SC + pn;
      accS[t][r] = v;
      rm[r] = fmaxf(rm[r], v);
    }
  }
#pragma unroll
  for (int msk = 1; msk < 16; msk <<= 1)
#pragma unroll
    for (int r = 0; r < 8; r++) rm[r] = fmaxf(rm[r], __shfl_xor(rm[r], msk, 32));
#pragma unroll
  for (int r = 0; r < 8; r++) rsum[r] = 0.f;
#pragma unroll
  for (int t = 0; t < 32; t++)
#pragma unroll
    for (int r = 0; r < 8; r++) {
      float e = __expf(accS[t][r] - rm[r]);
      accS[t][r] = e;
      rsum[r] += e;
    }
#pragma unroll
  for (int msk = 1; msk < 16; msk <<= 1)
#pragma unroll
    for (int r = 0; r < 8; r++) rsum[r] += __shfl_xor(rsum[r], msk, 32);
#pragma unroll
  for (int r = 0; r < 8; r++) rsum[r] = 1.f / rsum[r];
#pragma unroll
  for (int t = 0; t < 32; t++)
#pragma unroll
    for (int r = 0; r < 8; r++)
      P[(wave * 16 + r + 8 * half) * 512 + t * 16 + colid] = f2bf(accS[t][r] * rsum[r]);

  // ---- phase 2: O = P * V  (V staged transposed -> manual store path)
  v8f accO[8];
#pragma unroll
  for (int nt = 0; nt < 8; nt++) accO[nt] = vzero8();
#pragma unroll
  for (int ch = 0; ch < 8; ch++) {
    __syncthreads();
    {
      const unsigned short* vp = qkv + ((size_t)(b * 512 + ch * 64 + tid) * 768 + 512 + head * 128);
#pragma unroll
      for (int g = 0; g < 16; g++) {
        us8 v = *reinterpret_cast<const us8*>(vp + g * 8);
#pragma unroll
        for (int e = 0; e < 8; e++) KV[(g * 8 + e) * 64 + tid] = v[e];  // transposed: [d][k]
      }
    }
    __syncthreads();
#pragma unroll
    for (int ks2 = 0; ks2 < 2; ks2++) {
      Frag16 fa;
      const __bf16* pp = reinterpret_cast<const __bf16*>(P) + (wave * 16 + m) * 512 + ch * 64 + ks2 * 32 + kbA;
      fa.h[0] = *reinterpret_cast<const bf16x8*>(pp);
      fa.h[1] = *reinterpret_cast<const bf16x8*>(pp + 8);
#pragma unroll
      for (int nt = 0; nt < 8; nt++) {
        Frag16 fb;
        const __bf16* vb = reinterpret_cast<const __bf16*>(KV) + (nt * 16 + colid) * 64 + ks2 * 32 + kbB;
        fb.h[0] = *reinterpret_cast<const bf16x8*>(vb);
        fb.h[1] = *reinterpret_cast<const bf16x8*>(vb + 8);
        accO[nt] = wmma_bf16(fa.v, fb.v, accO[nt]);
      }
    }
  }
#pragma unroll
  for (int nt = 0; nt < 8; nt++)
#pragma unroll
    for (int r = 0; r < 8; r++) {
      int row = qtile * 32 + wave * 16 + r + 8 * half;
      int d = nt * 16 + colid;
      obuf[(size_t)(b * 512 + row) * 256 + head * 128 + d] = f2bf(accO[nt][r]);
    }
}

// ---------------------------------------------------------------- gather + pitch embed
__global__ __launch_bounds__(256) void k_gather(const int* __restrict__ mel2ph,
                                                const float* __restrict__ f0,
                                                const float* __restrict__ enc,
                                                const float* __restrict__ pitch_w,
                                                float* __restrict__ out) {
  int bf = blockIdx.x;  // b*4096 + frame
  int b = bf >> 12;
  int h = threadIdx.x;
  int mp = mel2ph[bf];
  float f0v = (mp > 0) ? f0[bf] : 0.f;
  float mel = 1127.0f * log1pf(f0v * (1.f / 700.f));
  float melmax = 1127.0f * log1pf(1100.f / 700.f);
  float melmin = 1127.0f * log1pf(50.f / 700.f);
  float a = 254.0f / (melmax - melmin);
  float bb = melmin * a - 1.0f;
  if (mel > 0.f) mel = mel * a - bb;
  mel = fminf(fmaxf(mel, 1.0f), 255.0f);
  int pit = (int)rintf(mel);
  float g = (mp > 0) ? enc[((size_t)(b * 512 + mp - 1)) * 256 + h] : 0.f;
  out[(size_t)bf * 256 + h] = g + pitch_w[pit * 256 + h];
}

// ================================================================ host
extern "C" void kernel_launch(void* const* d_in, const int* in_sizes, int n_in,
                              void* d_out, int out_size, void* d_ws, size_t ws_size,
                              hipStream_t stream) {
  const int*   tokens    = (const int*)d_in[0];
  const int*   durations = (const int*)d_in[1];
  const float* f0        = (const float*)d_in[2];
  const float* txt_w     = (const float*)d_in[4];
  const float* dur_w     = (const float*)d_in[5];
  const float* dur_b     = (const float*)d_in[6];
  const float* pitch_w   = (const float*)d_in[7];
  const float* Wqkv      = (const float*)d_in[8];
  const float* bqkv      = (const float*)d_in[9];
  const float* Wo        = (const float*)d_in[10];
  const float* bo        = (const float*)d_in[11];
  const float* ln1s      = (const float*)d_in[12];
  const float* ln1b      = (const float*)d_in[13];
  const float* Wffn1     = (const float*)d_in[14];
  const float* bffn1     = (const float*)d_in[15];
  const float* Wffn2     = (const float*)d_in[16];
  const float* bffn2     = (const float*)d_in[17];
  const float* ln2s      = (const float*)d_in[18];
  const float* ln2b      = (const float*)d_in[19];
  const float* lnfs      = (const float*)d_in[20];
  const float* lnfb      = (const float*)d_in[21];

  char* ws = (char*)d_ws;
  size_t off = 0;
  auto alloc = [&](size_t bytes) -> void* {
    void* p = ws + off;
    off = (off + bytes + 255) & ~(size_t)255;
    return p;
  };
  int* mel2ph            = (int*)alloc((size_t)16 * 4096 * 4);
  float* x               = (float*)alloc((size_t)8192 * 256 * 4);
  unsigned short* h2     = (unsigned short*)alloc((size_t)8192 * 256 * 2);
  unsigned short* qkvb   = (unsigned short*)alloc((size_t)8192 * 768 * 2);
  unsigned short* obuf   = (unsigned short*)alloc((size_t)8192 * 256 * 2);
  unsigned short* ffo    = (unsigned short*)alloc((size_t)8192 * 1024 * 2);
  float* enc             = (float*)alloc((size_t)8192 * 256 * 4);
  unsigned short* wqkv_b = (unsigned short*)alloc((size_t)4 * 256 * 768 * 2);
  unsigned short* wo_b   = (unsigned short*)alloc((size_t)4 * 256 * 256 * 2);
  unsigned short* wf1_b  = (unsigned short*)alloc((size_t)4 * 2304 * 1024 * 2);
  unsigned short* wf2_b  = (unsigned short*)alloc((size_t)4 * 1024 * 256 * 2);

  auto conv = [&](const float* src, unsigned short* dst, int n) {
    k_f32_to_bf16<<<(n + 255) / 256, 256, 0, stream>>>(src, dst, n);
  };
  conv(Wqkv, wqkv_b, 4 * 256 * 768);
  conv(Wo, wo_b, 4 * 256 * 256);
  conv(Wffn1, wf1_b, 4 * 2304 * 1024);
  conv(Wffn2, wf2_b, 4 * 1024 * 256);

  k_mel2ph<<<16, 512, 0, stream>>>(tokens, durations, mel2ph);
  k_embed<<<8192, 256, 0, stream>>>(tokens, durations, txt_w, dur_w, dur_b, x);

  for (int l = 0; l < 4; l++) {
    k_ln<true, false><<<1024, 256, 0, stream>>>(x, ln1s + l * 256, ln1b + l * 256, tokens, h2, nullptr);
    k_gemm<false, false, false, false, true><<<dim3(768 / 64, 8192 / 64), 256, 0, stream>>>(
        h2, wqkv_b + (size_t)l * 256 * 768, bqkv + l * 768, nullptr, tokens, qkvb, nullptr, 768, 256, 256);
    k_attn<<<dim3(16, 2, 16), 64, 0, stream>>>(qkvb, tokens, obuf);
    k_gemm<false, false, true, true, false><<<dim3(256 / 64, 8192 / 64), 256, 0, stream>>>(
        obuf, wo_b + (size_t)l * 256 * 256, bo + l * 256, x, tokens, nullptr, x, 256, 256, 256);
    k_ln<true, false><<<1024, 256, 0, stream>>>(x, ln2s + l * 256, ln2b + l * 256, tokens, h2, nullptr);
    k_gemm<true, true, false, false, true><<<dim3(1024 / 64, 8192 / 64), 256, 0, stream>>>(
        h2, wf1_b + (size_t)l * 2304 * 1024, bffn1 + l * 1024, nullptr, tokens, ffo, nullptr, 1024, 2304, 256);
    k_gemm<false, false, true, true, false><<<dim3(256 / 64, 8192 / 64), 256, 0, stream>>>(
        ffo, wf2_b + (size_t)l * 1024 * 256, bffn2 + l * 256, x, tokens, nullptr, x, 256, 1024, 1024);
  }
  k_ln<false, true><<<1024, 256, 0, stream>>>(x, lnfs, lnfb, tokens, nullptr, enc);
  k_gather<<<16 * 4096, 256, 0, stream>>>(mel2ph, f0, enc, pitch_w, (float*)d_out);
}
